// NeuralNetwork_88794153877504
// MI455X (gfx1250) — matrix operand order repaired and checked by reading, not compiled, not run on hardware
//
#include <hip/hip_runtime.h>
#include <hip/hip_bf16.h>
#include <math.h>

#define NSAMP 200
#define HASHMASK 524287u
#define P2H 2654435761u
#define P3H 805459861u

typedef __attribute__((ext_vector_type(16))) __bf16 v16bf;
typedef __attribute__((ext_vector_type(8)))  float  v8f;

// ---------- helpers ----------

__device__ __forceinline__ __bf16 f2bf(float f) {
    unsigned u = __float_as_uint(f);
    unsigned r = (u + 0x7FFFu + ((u >> 16) & 1u)) >> 16;   // round-to-nearest-even
    return __builtin_bit_cast(__bf16, (unsigned short)r);
}

__device__ __forceinline__ v8f wmma_bf16(v16bf a, v16bf b, v8f c) {
    return __builtin_amdgcn_wmma_f32_16x16x32_bf16(
        /*neg_a=*/false, a, /*neg_b=*/false, b,
        /*c_mod=*/(short)0, c, /*reuse_a=*/false, /*reuse_b=*/false);
}

// Load one pre-swizzled 16-bf16 fragment (32 bytes) -> 2x b128 loads.
__device__ __forceinline__ v16bf ldfrag(const __bf16* p) {
    return *(const v16bf*)p;
}

// One trilinear hash-grid level; tab pre-offset to this level, dim elements/entry.
__device__ __forceinline__ void grid_level(float x0, float x1, float x2,
                                           int res, bool dense,
                                           const float* __restrict__ tab,
                                           int dim, float* acc) {
    float p0 = x0 * (float)(res - 1);
    float p1 = x1 * (float)(res - 1);
    float p2 = x2 * (float)(res - 1);
    int g0 = (int)floorf(p0); g0 = g0 < 0 ? 0 : (g0 > res - 2 ? res - 2 : g0);
    int g1 = (int)floorf(p1); g1 = g1 < 0 ? 0 : (g1 > res - 2 ? res - 2 : g1);
    int g2 = (int)floorf(p2); g2 = g2 < 0 ? 0 : (g2 > res - 2 ? res - 2 : g2);
    float f0 = p0 - (float)g0, f1 = p1 - (float)g1, f2 = p2 - (float)g2;
    #pragma unroll
    for (int c = 0; c < 8; ++c) {
        int cx = (c >> 2) & 1, cy = (c >> 1) & 1, cz = c & 1;
        unsigned c0 = (unsigned)(g0 + cx), c1 = (unsigned)(g1 + cy), c2 = (unsigned)(g2 + cz);
        unsigned idx;
        if (dense) idx = (c0 * (unsigned)res + c1) * (unsigned)res + c2;
        else       idx = (c0 ^ (c1 * P2H) ^ (c2 * P3H)) & HASHMASK;
        float w = (cx ? f0 : 1.f - f0) * (cy ? f1 : 1.f - f1) * (cz ? f2 : 1.f - f2);
        const float* e = tab + (size_t)idx * dim;
        for (int j = 0; j < dim; ++j) acc[j] += w * e[j];
    }
}

// ---------- Kernel A: per-ray label field + reductions + selected RGB encode ----------

__global__ __launch_bounds__(256)
void ray_kernel(const float* __restrict__ x,
                const float* __restrict__ ltab,
                const float* __restrict__ lw, const float* __restrict__ lb,
                const float* __restrict__ rtab,
                float* __restrict__ out_hits, float* __restrict__ out_sigma,
                float* __restrict__ out_idx, __bf16* __restrict__ featfrag) {
    __shared__ unsigned s_max;
    __shared__ int s_first;
    __shared__ int s_sel;
    __shared__ float s_feat[32];

    const int ray = blockIdx.x;
    const int tid = threadIdx.x;
    if (tid == 0) { s_max = 0u; s_first = 0x7fffffff; }
    __syncthreads();

    const float* xr = x + (size_t)ray * 4;
    float th1 = xr[0], ph1 = xr[1], th2 = xr[2], ph2 = xr[3];
    float st1 = sinf(th1), ct1 = cosf(th1);
    float st2 = sinf(th2), ct2 = cosf(th2);
    float p1x = st1 * cosf(ph1), p1y = st1 * sinf(ph1), p1z = ct1;
    float p2x = st2 * cosf(ph2), p2y = st2 * sinf(ph2), p2z = ct2;
    float dx = p2x - p1x, dy = p2y - p1y, dz = p2z - p1z;

    static const int lres[6] = {8, 16, 32, 64, 128, 256};
    static const int loff[6] = {0, 512, 4608, 37376, 299520, 823808};

    if (tid < NSAMP) {
        float tt = (float)tid * (1.0f / (float)(NSAMP - 1));
        float px = p1x + dx * tt, py = p1y + dy * tt, pz = p1z + dz * tt;
        float x0 = fminf(fmaxf((px + 1.f) * 0.5f, 0.f), 1.f);
        float x1 = fminf(fmaxf((py + 1.f) * 0.5f, 0.f), 1.f);
        float x2 = fminf(fmaxf((pz + 1.f) * 0.5f, 0.f), 1.f);
        float z = lb[0];
        #pragma unroll
        for (int l = 0; l < 6; ++l) {
            float acc = 0.f;
            grid_level(x0, x1, x2, lres[l], l < 4, ltab + loff[l], 1, &acc);
            z += acc * lw[l];
        }
        float sigma = 1.f / (1.f + expf(-z));
        out_sigma[(size_t)ray * NSAMP + tid] = sigma;
        atomicMax(&s_max, __float_as_uint(sigma));      // sigma>0 -> bits monotonic
        if (sigma > 0.5f) atomicMin(&s_first, tid);
    }
    if (tid >= 23 && tid < 32) s_feat[tid] = 0.f;        // zero K-padding of feat
    __syncthreads();

    if (tid == 0) {
        out_hits[ray] = __uint_as_float(s_max);
        int first = (s_first == 0x7fffffff) ? 0 : s_first;
        int idx = first < 2 ? 2 : (first > NSAMP - 3 ? NSAMP - 3 : first);
        out_idx[ray] = (float)idx;
        s_sel = idx;
        float nrm = sqrtf(dx * dx + dy * dy + dz * dz);
        nrm = fminf(fmaxf(nrm, 1e-6f), 1e6f);
        s_feat[0] = dx / nrm; s_feat[1] = dy / nrm; s_feat[2] = dz / nrm;
    }
    __syncthreads();

    if (tid < 5) {   // 5 dense RGB levels at the single selected sample
        static const int rres[5] = {4, 8, 16, 32, 64};
        static const int roff[5] = {0, 64, 576, 4672, 37440};
        float tt = (float)s_sel * (1.0f / (float)(NSAMP - 1));
        float px = p1x + dx * tt, py = p1y + dy * tt, pz = p1z + dz * tt;
        float x0 = fminf(fmaxf((px + 1.f) * 0.5f, 0.f), 1.f);
        float x1 = fminf(fmaxf((py + 1.f) * 0.5f, 0.f), 1.f);
        float x2 = fminf(fmaxf((pz + 1.f) * 0.5f, 0.f), 1.f);
        float acc[4] = {0.f, 0.f, 0.f, 0.f};
        grid_level(x0, x1, x2, rres[tid], true, rtab + (size_t)roff[tid] * 4, 4, acc);
        s_feat[3 + tid * 4 + 0] = acc[0];
        s_feat[3 + tid * 4 + 1] = acc[1];
        s_feat[3 + tid * 4 + 2] = acc[2];
        s_feat[3 + tid * 4 + 3] = acc[3];
    }
    __syncthreads();

    // Emit feat row (K=32, bf16) directly in WMMA A-fragment layout:
    // group = ray>>4, row-in-group m = ray&15, K = tid.
    // laneA = m + 16*((K>>3)&1),  e = (K&7) + 8*((K>>4)&1)
    if (tid < 32) {
        int m = ray & 15, grp = ray >> 4;
        int laneA = m + 16 * ((tid >> 3) & 1);
        int e = (tid & 7) + 8 * ((tid >> 4) & 1);
        featfrag[(size_t)grp * 512 + laneA * 16 + e] = f2bf(s_feat[tid]);
    }
}

// ---------- Kernel B: fused MLP 32->64->64->64->3 via bf16 WMMA ----------
// All operands pre-swizzled into per-lane fragment order -> every fragment
// access is one 32-byte vector load (2x ds_load_b128 / global_load_b128).

// Scatter D tile (16x16 f32) with bias+ReLU into per-wave A-fragment bf16 store.
// hw points at this wave's activation region: [kb(2)][lane(32)][e(16)].
__device__ __forceinline__ void store_act(__bf16* hw, int nt, v8f c,
                                          const float* __restrict__ bias, int lane) {
    int hfd = lane >> 4;           // D layout: lanes 16-31 hold rows 8..15
    int nl  = lane & 15;
    int col = nt * 16 + nl;        // this is K of the next layer
    int kb = col >> 5, kw = col & 31;
    int laneA = 16 * ((kw >> 3) & 1);              // + m added per row below
    int e = (kw & 7) + 8 * ((kw >> 4) & 1);
    float bv = bias[col];
    #pragma unroll
    for (int r = 0; r < 8; ++r) {
        int m = r + 8 * hfd;       // row within the wave's 16-row tile
        float val = fmaxf(c[r] + bv, 0.f);
        hw[(kb * 32 + laneA + m) * 16 + e] = f2bf(val);
    }
}

__global__ __launch_bounds__(256)
void mlp_kernel(const __bf16* __restrict__ featfrag,
                const float* __restrict__ w1, const float* __restrict__ b1,
                const float* __restrict__ w2, const float* __restrict__ b2,
                const float* __restrict__ w3, const float* __restrict__ b3,
                const float* __restrict__ w4, const float* __restrict__ b4,
                float* __restrict__ out_rgb) {
    // Weight fragments: [kb][nt][lane][e] (B-fragment order: lane = n%16 + 16*((kw>>3)&1), e = (kw&7) + 8*((kw>>4)&1))
    __shared__ __align__(32) __bf16 swf1[4 * 32 * 16];        // K=32 (pad 23), N=64
    __shared__ __align__(32) __bf16 swf2[2 * 4 * 32 * 16];    // 64x64
    __shared__ __align__(32) __bf16 swf3[2 * 4 * 32 * 16];    // 64x64
    __shared__ __align__(32) __bf16 swf4[2 * 32 * 16];        // K=64, N=16 (pad 3)
    // Activations in A-fragment order, per wave: [wid][kb(2)][lane][e]
    __shared__ __align__(32) __bf16 hfA[8 * 2 * 32 * 16];
    __shared__ __align__(32) __bf16 hfB[8 * 2 * 32 * 16];

    const int t = threadIdx.x;
    for (int i = t; i < 32 * 64; i += 256) {
        int k = i >> 6, n = i & 63;
        float v = (k < 23) ? w1[k * 64 + n] : 0.f;
        int hf = (k >> 3) & 1;
        swf1[(((n >> 4) * 32) + (n & 15) + 16 * hf) * 16 + ((k & 7) + 8 * ((k >> 4) & 1))] = f2bf(v);
    }
    for (int i = t; i < 64 * 64; i += 256) {
        int k = i >> 6, n = i & 63;
        int kb = k >> 5, kw = k & 31, hf = (kw >> 3) & 1;
        int idx = (((kb * 4 + (n >> 4)) * 32) + (n & 15) + 16 * hf) * 16 + ((kw & 7) + 8 * ((kw >> 4) & 1));
        swf2[idx] = f2bf(w2[i]);
        swf3[idx] = f2bf(w3[i]);
    }
    for (int i = t; i < 64 * 16; i += 256) {
        int k = i >> 4, n = i & 15;
        float v = (n < 3) ? w4[k * 3 + n] : 0.f;
        int kb = k >> 5, kw = k & 31, hf = (kw >> 3) & 1;
        swf4[((kb * 32) + n + 16 * hf) * 16 + ((kw & 7) + 8 * ((kw >> 4) & 1))] = f2bf(v);
    }
    __syncthreads();

    const int lane = t & 31, wid = t >> 5;
    const int rowG = blockIdx.x * 128 + wid * 16;   // global row base of this wave
    const __bf16* gA = featfrag + ((size_t)(blockIdx.x * 8 + wid)) * 512;
    __bf16* hA = hfA + wid * 1024;
    __bf16* hB = hfB + wid * 1024;

    // ---- layer 1: feat (K=32) @ w1 -> hA ----
    {
        v16bf a = ldfrag(gA + lane * 16);
        #pragma unroll
        for (int nt = 0; nt < 4; ++nt) {
            v8f c = {};
            c = wmma_bf16(a, ldfrag(&swf1[(nt * 32 + lane) * 16]), c);
            store_act(hA, nt, c, b1, lane);
        }
    }
    __syncthreads();
    // ---- layer 2: hA @ w2 -> hB ----
    {
        v16bf a0 = ldfrag(hA + lane * 16);
        v16bf a1 = ldfrag(hA + 512 + lane * 16);
        #pragma unroll
        for (int nt = 0; nt < 4; ++nt) {
            v8f c = {};
            c = wmma_bf16(a0, ldfrag(&swf2[((0 * 4 + nt) * 32 + lane) * 16]), c);
            c = wmma_bf16(a1, ldfrag(&swf2[((1 * 4 + nt) * 32 + lane) * 16]), c);
            store_act(hB, nt, c, b2, lane);
        }
    }
    __syncthreads();
    // ---- layer 3: hB @ w3 -> hA ----
    {
        v16bf a0 = ldfrag(hB + lane * 16);
        v16bf a1 = ldfrag(hB + 512 + lane * 16);
        #pragma unroll
        for (int nt = 0; nt < 4; ++nt) {
            v8f c = {};
            c = wmma_bf16(a0, ldfrag(&swf3[((0 * 4 + nt) * 32 + lane) * 16]), c);
            c = wmma_bf16(a1, ldfrag(&swf3[((1 * 4 + nt) * 32 + lane) * 16]), c);
            store_act(hA, nt, c, b3, lane);
        }
    }
    __syncthreads();
    // ---- layer 4: hA @ w4 (64 -> 3, N padded to 16) ----
    {
        v16bf a0 = ldfrag(hA + lane * 16);
        v16bf a1 = ldfrag(hA + 512 + lane * 16);
        v8f c = {};
        c = wmma_bf16(a0, ldfrag(&swf4[(0 * 32 + lane) * 16]), c);
        c = wmma_bf16(a1, ldfrag(&swf4[(1 * 32 + lane) * 16]), c);
        int hf = lane >> 4;
        int col = lane & 15;
        if (col < 3) {
            float bv = b4[col];
            #pragma unroll
            for (int r = 0; r < 8; ++r) {
                int row = rowG + r + 8 * hf;
                out_rgb[(size_t)row * 3 + col] = c[r] + bv;
            }
        }
    }
}

// ---------- launch ----------

extern "C" void kernel_launch(void* const* d_in, const int* in_sizes, int n_in,
                              void* d_out, int out_size, void* d_ws, size_t ws_size,
                              hipStream_t stream) {
    const float* x    = (const float*)d_in[0];
    const float* ltab = (const float*)d_in[1];
    const float* lw   = (const float*)d_in[2];
    const float* lb   = (const float*)d_in[3];
    const float* rtab = (const float*)d_in[4];
    const float* w1   = (const float*)d_in[5];
    const float* b1   = (const float*)d_in[6];
    const float* w2   = (const float*)d_in[7];
    const float* b2   = (const float*)d_in[8];
    const float* w3   = (const float*)d_in[9];
    const float* b3   = (const float*)d_in[10];
    const float* w4   = (const float*)d_in[11];
    const float* b4   = (const float*)d_in[12];

    const int N = in_sizes[0] / 4;                 // 32768 rays

    float* out_hits  = (float*)d_out;              // [N]
    float* out_sigma = out_hits + N;               // [N,200]
    float* out_idx   = out_sigma + (size_t)N * NSAMP;  // [N]
    float* out_rgb   = out_idx + N;                // [N,3]

    __bf16* featfrag = (__bf16*)d_ws;              // [N/16][32 lanes][16] bf16 A-fragments

    ray_kernel<<<N, 256, 0, stream>>>(x, ltab, lw, lb, rtab,
                                      out_hits, out_sigma, out_idx, featfrag);
    mlp_kernel<<<N / 128, 256, 0, stream>>>(featfrag, w1, b1, w2, b2, w3, b3, w4, b4,
                                            out_rgb);
}
